// ConceptualMambaBlock_30288109371860
// MI455X (gfx1250) — matrix operand order repaired and checked by reading, not compiled, not run on hardware
//
#include <hip/hip_runtime.h>
#include <hip/hip_bf16.h>
#include <hip/hip_fp16.h>

// Fused ConceptualMambaBlock for MI455X (gfx1250, wave32, WMMA).
//
// grid = B * (N/16) = 128 workgroups, one (b, n-tile) each; 8 waves/WG.
// Per WG: weights live in LDS as f16 hi/lo; t-loop (sequential EMA scan):
//   stage x-tile (software-pipelined global loads) -> LDS (hi/lo f16)
//   -> GEMM1 (3-term f16 WMMA, f32 acc)
//   -> bias + EMA (state in v8f regs) + sigmoid + gate -> LDS transpose
//   -> GEMM2 (3-term f16 WMMA, W_con fragments hoisted in VGPRs) -> bias -> store.
//
// t-loop is pinned to unroll(1): iterations are serialized by barriers anyway,
// and x2 unroll pushed VGPRs past 256 (s_set_vgpr_msb overhead in round 2).

typedef _Float16 v16h __attribute__((ext_vector_type(16)));
typedef float    v8f  __attribute__((ext_vector_type(8)));

namespace {

constexpr int Bdim = 4;
constexpr int Tdim = 96;
constexpr int Ndim = 512;
constexpr int Hdim = 128;
constexpr int Odim = 256;   // 2*H

union FragU { v16h f; uint4 q[2]; };

__device__ __forceinline__ v8f wmma_acc(v16h a, v16h b, v8f c) {
  // D = A(16x32 f16) * B(32x16 f16) + C(16x16 f32)
  return __builtin_amdgcn_wmma_f32_16x16x32_f16(
      /*neg_a=*/false, a, /*neg_b=*/false, b,
      /*c_mod=*/(short)0, c, /*reuse_a=*/false, /*reuse_b=*/false);
}

// A fragment (16x32): row = lane&15; halves j<8 -> K=kb+j, j>=8 -> K=kb+16+(j-8),
// kb = (lane>>4)*8.  Source LDS layout: row-major [row][128] halves.
__device__ __forceinline__ v16h loadA(const _Float16* base, int lane, int hb) {
  const _Float16* p = base + (lane & 15) * Hdim + hb + ((lane >> 4) << 3);
  FragU u;
  u.q[0] = *(const uint4*)(p);        // K = kb .. kb+7
  u.q[1] = *(const uint4*)(p + 16);   // K = kb+16 .. kb+23
  return u.f;
}

// B fragment (32x16): col = lane&15 (maps to weight row `o`); halves j<8 -> K=kb+j, j>=8 -> K=kb+16+(j-8), kb = (lane>>4)*8.
// Source LDS layout: row-major [o][128] halves (B[k][n] = W[o][h] with n<->o, k<->h).
__device__ __forceinline__ v16h loadB(const _Float16* base, int o, int hb, int lane) {
  const _Float16* p = base + o * Hdim + hb + ((lane >> 4) << 3);
  FragU u;
  u.q[0] = *(const uint4*)(p);        // K = kb .. kb+7
  u.q[1] = *(const uint4*)(p + 16);   // K = kb+16 .. kb+23
  return u.f;
}

__device__ __forceinline__ void split_f32(float v, _Float16* hi, _Float16* lo) {
  _Float16 h = (_Float16)v;
  *hi = h;
  *lo = (_Float16)(v - (float)h);
}

} // namespace

__global__ __launch_bounds__(256)
void mamba_block_fused(const float* __restrict__ x,
                       const float* __restrict__ Wexp,
                       const float* __restrict__ bexp,
                       const float* __restrict__ Wcon,
                       const float* __restrict__ bcon,
                       float* __restrict__ out)
{
  // ~210 KB LDS (<= 320 KB per WGP on CDNA5); one workgroup per WGP.
  __shared__ _Float16 sW1h[Odim * Hdim];
  __shared__ _Float16 sW1l[Odim * Hdim];
  __shared__ _Float16 sW2h[Hdim * Hdim];
  __shared__ _Float16 sW2l[Hdim * Hdim];
  __shared__ _Float16 sXh[16 * Hdim];
  __shared__ _Float16 sXl[16 * Hdim];
  __shared__ _Float16 sYh[16 * Hdim];
  __shared__ _Float16 sYl[16 * Hdim];
  __shared__ float    sbe[Odim];
  __shared__ float    sbc[Hdim];

  const int tid  = threadIdx.x;
  const int lane = tid & 31;
  const int wv   = tid >> 5;             // wave id 0..7
  const int col  = lane & 15;            // C/D fragment column
  const int mhi  = (lane >> 4) << 3;     // C/D fragment row base (0 or 8)

  const int b  = blockIdx.x >> 5;        // batch 0..3
  const int n0 = (blockIdx.x & 31) << 4; // n-tile origin

  // ---- one-time: weights + bias -> LDS as f16 hi/lo ----
  for (int i = tid; i < Odim * Hdim / 4; i += 256) {
    float4 v = ((const float4*)Wexp)[i];
    int o4 = i * 4;
    split_f32(v.x, &sW1h[o4 + 0], &sW1l[o4 + 0]);
    split_f32(v.y, &sW1h[o4 + 1], &sW1l[o4 + 1]);
    split_f32(v.z, &sW1h[o4 + 2], &sW1l[o4 + 2]);
    split_f32(v.w, &sW1h[o4 + 3], &sW1l[o4 + 3]);
  }
  for (int i = tid; i < Hdim * Hdim / 4; i += 256) {
    float4 v = ((const float4*)Wcon)[i];
    int o4 = i * 4;
    split_f32(v.x, &sW2h[o4 + 0], &sW2l[o4 + 0]);
    split_f32(v.y, &sW2h[o4 + 1], &sW2l[o4 + 1]);
    split_f32(v.z, &sW2h[o4 + 2], &sW2l[o4 + 2]);
    split_f32(v.w, &sW2h[o4 + 3], &sW2l[o4 + 3]);
  }
  if (tid < Odim) sbe[tid] = bexp[tid];
  if (tid < Hdim) sbc[tid] = bcon[tid];
  __syncthreads();

  const size_t plane = (size_t)Ndim * Hdim;                   // one (b,t) slab
  const float* xb = x   + (size_t)b * Tdim * plane + (size_t)n0 * Hdim;
  float*       ob = out + (size_t)b * Tdim * plane + (size_t)n0 * Hdim;

  // per-wave bias scalars (column-indexed, uniform across fragment rows)
  const float biasP = sbe[wv * 16 + col];          // primary channel h = 16w+col
  const float biasG = sbe[128 + wv * 16 + col];    // paired gating channel
  const float bias2 = sbc[wv * 16 + col];

  const int oP = wv * 16 + col;          // W_exp row for primary tile
  const int oG = 128 + wv * 16 + col;    // W_exp row for gating tile
  const int oC = wv * 16 + col;          // W_con row for output tile

  // Hoist GEMM2 B-fragments (t-invariant): 8 x v16h = 64 VGPRs.
  v16h bC_h[4], bC_l[4];
#pragma unroll
  for (int ks = 0; ks < 4; ++ks) {
    bC_h[ks] = loadB(sW2h, oC, ks * 32, lane);
    bC_l[ks] = loadB(sW2l, oC, ks * 32, lane);
  }

  // EMA state fragment: element r <-> (n = n0+mhi+r, h = 16*wv+col); persists over t.
  v8f st = {0.f, 0.f, 0.f, 0.f, 0.f, 0.f, 0.f, 0.f};

  // Software pipeline: x(t) staged in registers one iteration ahead so the
  // global-load latency is hidden behind the previous timestep's GEMMs.
  float4 xv0 = ((const float4*)xb)[tid];
  float4 xv1 = ((const float4*)xb)[tid + 256];

#pragma unroll 1
  for (int t = 0; t < Tdim; ++t) {
    // ---- write staged x(t) regs into LDS as hi/lo f16 ----
    {
      int o4 = tid * 4;
      split_f32(xv0.x, &sXh[o4 + 0], &sXl[o4 + 0]);
      split_f32(xv0.y, &sXh[o4 + 1], &sXl[o4 + 1]);
      split_f32(xv0.z, &sXh[o4 + 2], &sXl[o4 + 2]);
      split_f32(xv0.w, &sXh[o4 + 3], &sXl[o4 + 3]);
      int o5 = (tid + 256) * 4;
      split_f32(xv1.x, &sXh[o5 + 0], &sXl[o5 + 0]);
      split_f32(xv1.y, &sXh[o5 + 1], &sXl[o5 + 1]);
      split_f32(xv1.z, &sXh[o5 + 2], &sXl[o5 + 2]);
      split_f32(xv1.w, &sXh[o5 + 3], &sXl[o5 + 3]);
    }
    __syncthreads();

    // ---- issue next timestep's global loads immediately (latency hiding) ----
    if (t + 1 < Tdim) {
      const float* xn = xb + (size_t)(t + 1) * plane;
      xv0 = ((const float4*)xn)[tid];
      xv1 = ((const float4*)xn)[tid + 256];
    }

    // ---- GEMM1: wave computes primary tile wv and gating tile 8+wv ----
    v8f accP = {0.f, 0.f, 0.f, 0.f, 0.f, 0.f, 0.f, 0.f};
    v8f accG = {0.f, 0.f, 0.f, 0.f, 0.f, 0.f, 0.f, 0.f};
#pragma unroll
    for (int ks = 0; ks < 4; ++ks) {
      const int hb = ks * 32;
      v16h ah  = loadA(sXh, lane, hb);
      v16h al  = loadA(sXl, lane, hb);
      v16h bph = loadB(sW1h, oP, hb, lane);
      v16h bpl = loadB(sW1l, oP, hb, lane);
      v16h bgh = loadB(sW1h, oG, hb, lane);
      v16h bgl = loadB(sW1l, oG, hb, lane);
      // 3-term error-compensated f16 product (~fp32 accuracy)
      accP = wmma_acc(ah, bph, accP);
      accP = wmma_acc(ah, bpl, accP);
      accP = wmma_acc(al, bph, accP);
      accG = wmma_acc(ah, bgh, accG);
      accG = wmma_acc(ah, bgl, accG);
      accG = wmma_acc(al, bgh, accG);
    }

    // ---- bias + EMA scan step + sigmoid + gate; transpose via LDS ----
#pragma unroll
    for (int r = 0; r < 8; ++r) {
      float g = accG[r] + biasG;
      float s = st[r] * 0.9f + g * 0.1f;   // EMA: s_t = 0.9 s_{t-1} + 0.1 g_t
      st[r] = s;
      float sig = 1.0f / (1.0f + __expf(-s));
      float a = (accP[r] + biasP) * sig;
      int idx = (mhi + r) * Hdim + (wv * 16 + col);   // [n_local][h] layout
      split_f32(a, &sYh[idx], &sYl[idx]);
    }
    __syncthreads();

    // ---- GEMM2: wave computes output tile wv (o in [16w,16w+16)) ----
    v8f acc2 = {0.f, 0.f, 0.f, 0.f, 0.f, 0.f, 0.f, 0.f};
#pragma unroll
    for (int ks = 0; ks < 4; ++ks) {
      const int hb = ks * 32;
      v16h ah = loadA(sYh, lane, hb);
      v16h al = loadA(sYl, lane, hb);
      acc2 = wmma_acc(ah, bC_h[ks], acc2);
      acc2 = wmma_acc(ah, bC_l[ks], acc2);
      acc2 = wmma_acc(al, bC_h[ks], acc2);
    }

    float* ot = ob + (size_t)t * plane;
#pragma unroll
    for (int r = 0; r < 8; ++r) {
      ot[(size_t)(mhi + r) * Hdim + oC] = acc2[r] + bias2;
    }
    // No end-of-loop barrier needed: the next iteration's __syncthreads()
    // (after the sX restage, which touches neither sY nor the hoisted W_con
    // fragments) orders sY reuse; every wave's last sX read precedes its sY
    // write, which precedes that barrier.
  }
}

extern "C" void kernel_launch(void* const* d_in, const int* in_sizes, int n_in,
                              void* d_out, int out_size, void* d_ws, size_t ws_size,
                              hipStream_t stream) {
  (void)in_sizes; (void)n_in; (void)out_size; (void)d_ws; (void)ws_size;
  const float* x    = (const float*)d_in[0];
  const float* Wexp = (const float*)d_in[1];
  const float* bexp = (const float*)d_in[2];
  const float* Wcon = (const float*)d_in[3];
  const float* bcon = (const float*)d_in[4];
  float* out = (float*)d_out;

  dim3 grid(Bdim * (Ndim / 16));   // 128 workgroups: one per (b, 16-wide n-tile)
  dim3 block(256);                 // 8 wave32 waves
  hipLaunchKernelGGL(mamba_block_fused, grid, block, 0, stream,
                     x, Wexp, bexp, Wcon, bcon, out);
}
